// ChessStructureAttention_76373108457569
// MI455X (gfx1250) — matrix operand order repaired and checked by reading, not compiled, not run on hardware
//
#include <hip/hip_runtime.h>

typedef _Float16 f16;
typedef __attribute__((ext_vector_type(16))) _Float16 v16h;
typedef __attribute__((ext_vector_type(8)))  _Float16 v8h;
typedef __attribute__((ext_vector_type(4)))  _Float16 v4h;
typedef __attribute__((ext_vector_type(8)))  float    v8f;

#define DIM 512
#define NH  8
#define SQ  64
#define DH  64
#define NB  1024
#define KT  (DIM / 32)          // 16 k-steps of 32

#define WMMA(A,B,C) __builtin_amdgcn_wmma_f32_16x16x32_f16(false,(A),false,(B),(short)0,(C),false,false)

__device__ __forceinline__ v16h loadA16(const f16* __restrict__ row, int kt, int lh) {
  const int d0 = kt * 32 + 16 * lh;
  v8h a0 = *(const v8h*)(row + d0);
  v8h a1 = *(const v8h*)(row + d0 + 8);
  return __builtin_shufflevector(a0, a1, 0,1,2,3,4,5,6,7,8,9,10,11,12,13,14,15);
}

// ---------------------------------------------------------------- cvt f32->f16
__global__ __launch_bounds__(256) void cvt_f32_f16(const float* __restrict__ x,
                                                   f16* __restrict__ o, int n4) {
  int i = blockIdx.x * 256 + threadIdx.x;
  if (i < n4) {
    float4 v = ((const float4*)x)[i];
    v4h t;
    t[0] = (f16)v.x; t[1] = (f16)v.y; t[2] = (f16)v.z; t[3] = (f16)v.w;
    ((v4h*)o)[i] = t;
  }
}

// --------------------------------------------------- pack weight into B layout
// pack[((ntile*KT + kt)*32 + lane)*16 + j] = W[k][n]
//   n = ntile*16 + lane%16 ;  k = kt*32 + 16*(lane/16) + j
__global__ __launch_bounds__(256) void pack_w(const float* __restrict__ W,
                                              f16* __restrict__ out) {
  int idx   = blockIdx.x * 256 + threadIdx.x;     // DIM*DIM elements
  int j     = idx & 15;
  int L     = (idx >> 4) & 31;
  int kt    = (idx >> 9) & (KT - 1);
  int ntile = idx >> 13;
  int n = ntile * 16 + (L & 15);
  int k = kt * 32 + 16 * (L >> 4) + j;
  out[idx] = (f16)W[(size_t)k * DIM + n];
}

// ------------------------------------------------------------------- GEMM f16
// One wave -> 32 rows x 64 cols (8 WMMAs / k-step), register double-buffered.
// mode 0: f16 head layout [b,h,s,d]  (Q,K)   - LDS-staged coalesced stores
// mode 1: f16 transposed  [b,h,d,t]  (V)     - direct v8h stores (t contiguous)
// mode 2: f32 row-major   [m,DIM]    (out)   - LDS-staged coalesced stores
__global__ __launch_bounds__(128)
void gemm_f16(const f16* __restrict__ A, const f16* __restrict__ Wp,
              const float* __restrict__ bias, f16* __restrict__ o16,
              float* __restrict__ o32, int mode) {
  const int lane = threadIdx.x & 31;
  const int wave = threadIdx.x >> 5;
  const int ln16 = lane & 15;
  const int lh   = lane >> 4;
  const int mtile = blockIdx.x;                  // 32 rows
  const int nstrip = blockIdx.y * 4 + wave;      // 64-col strip (== head idx)
  const int colbase = nstrip * 64;
  const int ntile_base = nstrip * 4;

  __shared__ __align__(16) float sstage[4][16 * 64];   // 16 KB, per-wave stage

  const f16* arow0 = A + ((size_t)mtile * 32 + ln16) * DIM;
  const f16* arow1 = arow0 + (size_t)16 * DIM;
  const f16* wbase = Wp + (((size_t)ntile_base * KT) * 32 + lane) * 16;
  // B tile i at k-step kt: wbase + i*KT*512 + kt*512   (halves)

  v16h a0 = loadA16(arow0, 0, lh);
  v16h a1 = loadA16(arow1, 0, lh);
  v16h b0 = *(const v16h*)(wbase + 0 * KT * 512);
  v16h b1 = *(const v16h*)(wbase + 1 * KT * 512);
  v16h b2 = *(const v16h*)(wbase + 2 * KT * 512);
  v16h b3 = *(const v16h*)(wbase + 3 * KT * 512);

  v8f acc[8] = {};
#pragma unroll
  for (int kt = 0; kt < KT; ++kt) {
    const int kn = (kt + 1 < KT) ? kt + 1 : kt;
    v16h na0 = loadA16(arow0, kn, lh);
    v16h na1 = loadA16(arow1, kn, lh);
    v16h nb0 = *(const v16h*)(wbase + 0 * KT * 512 + kn * 512);
    v16h nb1 = *(const v16h*)(wbase + 1 * KT * 512 + kn * 512);
    v16h nb2 = *(const v16h*)(wbase + 2 * KT * 512 + kn * 512);
    v16h nb3 = *(const v16h*)(wbase + 3 * KT * 512 + kn * 512);
    acc[0] = WMMA(a0, b0, acc[0]);
    acc[1] = WMMA(a0, b1, acc[1]);
    acc[2] = WMMA(a0, b2, acc[2]);
    acc[3] = WMMA(a0, b3, acc[3]);
    acc[4] = WMMA(a1, b0, acc[4]);
    acc[5] = WMMA(a1, b1, acc[5]);
    acc[6] = WMMA(a1, b2, acc[6]);
    acc[7] = WMMA(a1, b3, acc[7]);
    a0 = na0; a1 = na1; b0 = nb0; b1 = nb1; b2 = nb2; b3 = nb3;
  }

  float bv4[4];
  for (int i = 0; i < 4; ++i) bv4[i] = bias[colbase + i * 16 + ln16];
  const int hh = nstrip;                       // head index for modes 0/1

  for (int mt = 0; mt < 2; ++mt) {
    const int mrow0 = mtile * 32 + mt * 16;
    if (mode == 1) {
      const int m0 = mrow0 + 8 * lh;
      const int bb = m0 >> 6, t0 = m0 & 63;
      for (int i = 0; i < 4; ++i) {
        const int d = i * 16 + ln16;
        v8h pk;
        for (int r = 0; r < 8; ++r) pk[r] = (f16)(acc[mt * 4 + i][r] + bv4[i]);
        *(v8h*)(o16 + (((size_t)bb * NH + hh) * DH + d) * SQ + t0) = pk;
      }
    } else if (mode == 0) {
      f16* stg = (f16*)(&sstage[wave][0]);
      for (int i = 0; i < 4; ++i)
        for (int r = 0; r < 8; ++r)
          stg[(r + 8 * lh) * 64 + i * 16 + ln16] = (f16)(acc[mt * 4 + i][r] + bv4[i]);
      const int m = mrow0 + ln16;
      const int bb = m >> 6, s = m & 63;
      const f16* src = stg + ln16 * 64 + lh * 32;
      f16* dst = o16 + (((size_t)bb * NH + hh) * SQ + s) * DH + lh * 32;
      *(v8h*)(dst)      = *(const v8h*)(src);
      *(v8h*)(dst + 8)  = *(const v8h*)(src + 8);
      *(v8h*)(dst + 16) = *(const v8h*)(src + 16);
      *(v8h*)(dst + 24) = *(const v8h*)(src + 24);
    } else {
      float* stg = &sstage[wave][0];
      for (int i = 0; i < 4; ++i)
        for (int r = 0; r < 8; ++r)
          stg[(r + 8 * lh) * 64 + i * 16 + ln16] = acc[mt * 4 + i][r] + bv4[i];
      const int m = mrow0 + ln16;
      const float* src = stg + ln16 * 64 + lh * 32;
      float* dst = o32 + (size_t)m * DIM + colbase + lh * 32;
      for (int q = 0; q < 8; ++q)
        ((float4*)dst)[q] = ((const float4*)src)[q];
    }
  }
}

// ------------------------------------------------------------------ attention
// One block per (b,h): 4 waves x 16 rows. QK^T -> bias/mask/softmax -> PV.
__global__ __launch_bounds__(128)
void attn_kernel(const f16* __restrict__ qh, const f16* __restrict__ kh,
                 const f16* __restrict__ vt, const unsigned char* __restrict__ hm,
                 const float* __restrict__ rel_bias, f16* __restrict__ ctx) {
  const int bh = blockIdx.x;
  const int b  = bh / NH;
  const int h  = bh % NH;
  const int lane = threadIdx.x & 31;
  const int wave = threadIdx.x >> 5;
  const int ln16 = lane & 15;
  const int lh   = lane >> 4;

  __shared__ float sbias[225];
  __shared__ __align__(16) f16 sp[4][16][SQ];          // P stage, reused for ctx
  __shared__ __align__(16) unsigned char smask[SQ * SQ];

  for (int i = threadIdx.x; i < 225; i += 128)
    sbias[i] = rel_bias[h * 225 + i];
  {
    const uint4* msrc = (const uint4*)(hm + (size_t)bh * SQ * SQ);
    uint4* mdst = (uint4*)smask;
    for (int i = threadIdx.x; i < (SQ * SQ / 16); i += 128)
      mdst[i] = msrc[i];
  }
  __syncthreads();

  const f16* qbase = qh + (size_t)bh * SQ * DH;
  const f16* kbase = kh + (size_t)bh * SQ * DH;
  const f16* vbase = vt + (size_t)bh * DH * SQ;

  // A operand from Q: row s = wave*16 + ln16
  const f16* qrow = qbase + (size_t)(wave * 16 + ln16) * DH;
  v16h aq0 = loadA16(qrow, 0, lh);
  v16h aq1 = loadA16(qrow, 1, lh);

  // preload all 8 B operands for QK^T, then run the WMMA chain
  v16h kb[8];
  for (int tt = 0; tt < 4; ++tt)
    for (int kt = 0; kt < 2; ++kt)
      kb[tt * 2 + kt] =
          *(const v16h*)(kbase + (size_t)(tt * 16 + ln16) * DH + kt * 32 + 16 * lh);

  v8f c[4] = {};
  for (int tt = 0; tt < 4; ++tt) {
    c[tt] = WMMA(aq0, kb[tt * 2 + 0], c[tt]);
    c[tt] = WMMA(aq1, kb[tt * 2 + 1], c[tt]);
  }

  // scale + relative bias + mask (mask from LDS)
  const float scale = 0.125f;   // 1/sqrt(64)
  float vals[4][8];
  for (int tt = 0; tt < 4; ++tt) {
    const int t = tt * 16 + ln16;
    const int rt = t >> 3, ft = t & 7;
    for (int r = 0; r < 8; ++r) {
      const int s = wave * 16 + r + 8 * lh;
      const int rs = s >> 3, fs = s & 7;
      float v = c[tt][r] * scale + sbias[(rs - rt + 7) * 15 + (fs - ft + 7)];
      v = smask[s * SQ + t] ? v : -3.0e38f;
      vals[tt][r] = v;
    }
  }

  // row softmax: rows live in 16-lane groups -> width-16 xor reductions
  for (int r = 0; r < 8; ++r) {
    float m = vals[0][r];
    m = fmaxf(m, vals[1][r]); m = fmaxf(m, vals[2][r]); m = fmaxf(m, vals[3][r]);
    for (int off = 1; off < 16; off <<= 1) m = fmaxf(m, __shfl_xor(m, off, 16));
    float ssum = 0.f;
    for (int tt = 0; tt < 4; ++tt) {
      float e = __expf(vals[tt][r] - m);
      vals[tt][r] = e;
      ssum += e;
    }
    for (int off = 1; off < 16; off <<= 1) ssum += __shfl_xor(ssum, off, 16);
    const float inv = 1.0f / ssum;
    const int srow = r + 8 * lh;
    for (int tt = 0; tt < 4; ++tt)
      sp[wave][srow][tt * 16 + ln16] = (f16)(vals[tt][r] * inv);
  }

  // P @ V : A operand of P re-read from LDS in A layout
  v16h ap0, ap1;
  {
    const f16* prow = &sp[wave][ln16][0];
    v8h p0 = *(const v8h*)(prow + 16 * lh);
    v8h p1 = *(const v8h*)(prow + 16 * lh + 8);
    ap0 = __builtin_shufflevector(p0, p1, 0,1,2,3,4,5,6,7,8,9,10,11,12,13,14,15);
    v8h p2 = *(const v8h*)(prow + 32 + 16 * lh);
    v8h p3 = *(const v8h*)(prow + 32 + 16 * lh + 8);
    ap1 = __builtin_shufflevector(p2, p3, 0,1,2,3,4,5,6,7,8,9,10,11,12,13,14,15);
  }

  v16h vb[8];
  for (int dt = 0; dt < 4; ++dt)
    for (int kt = 0; kt < 2; ++kt)
      vb[dt * 2 + kt] =
          *(const v16h*)(vbase + (size_t)(dt * 16 + ln16) * SQ + kt * 32 + 16 * lh);

  v8f oc[4] = {};
  for (int dt = 0; dt < 4; ++dt) {
    oc[dt] = WMMA(ap0, vb[dt * 2 + 0], oc[dt]);
    oc[dt] = WMMA(ap1, vb[dt * 2 + 1], oc[dt]);
  }

  // stage ctx tile in LDS (reuse sp, wave-local in-order), store coalesced rows
  for (int dt = 0; dt < 4; ++dt)
    for (int r = 0; r < 8; ++r)
      sp[wave][r + 8 * lh][dt * 16 + ln16] = (f16)oc[dt][r];
  {
    const int s = wave * 16 + ln16;
    const f16* src = &sp[wave][ln16][lh * 32];
    f16* dst = ctx + ((size_t)(b * SQ + s)) * DIM + h * DH + lh * 32;
    *(v8h*)(dst)      = *(const v8h*)(src);
    *(v8h*)(dst + 8)  = *(const v8h*)(src + 8);
    *(v8h*)(dst + 16) = *(const v8h*)(src + 16);
    *(v8h*)(dst + 24) = *(const v8h*)(src + 24);
  }
}

// ------------------------------------------------------------------- launcher
extern "C" void kernel_launch(void* const* d_in, const int* in_sizes, int n_in,
                              void* d_out, int out_size, void* d_ws, size_t ws_size,
                              hipStream_t stream) {
  const float* x   = (const float*)d_in[0];
  const unsigned char* hm = (const unsigned char*)d_in[1];   // jnp bool -> 1B
  const float* Wq  = (const float*)d_in[2];
  const float* bq  = (const float*)d_in[3];
  const float* Wk  = (const float*)d_in[4];
  const float* bk  = (const float*)d_in[5];
  const float* Wv  = (const float*)d_in[6];
  const float* bvp = (const float*)d_in[7];
  const float* Wo  = (const float*)d_in[8];
  const float* bo  = (const float*)d_in[9];
  const float* rb  = (const float*)d_in[10];
  float* out = (float*)d_out;

  const size_t NX = (size_t)NB * SQ * DIM;     // 33,554,432 elements
  f16* xh  = (f16*)d_ws;
  f16* qh  = xh  + NX;
  f16* kh  = qh  + NX;
  f16* vt  = kh  + NX;
  f16* ctx = vt  + NX;
  f16* wqp = ctx + NX;
  f16* wkp = wqp + (size_t)DIM * DIM;
  f16* wvp = wkp + (size_t)DIM * DIM;
  f16* wop = wvp + (size_t)DIM * DIM;

  // 1. x -> f16
  {
    int n4 = (int)(NX / 4);
    cvt_f32_f16<<<n4 / 256, 256, 0, stream>>>(x, xh, n4);
  }
  // 2. weight repack to WMMA B layout
  {
    int blocks = (DIM * DIM) / 256;
    pack_w<<<blocks, 256, 0, stream>>>(Wq, wqp);
    pack_w<<<blocks, 256, 0, stream>>>(Wk, wkp);
    pack_w<<<blocks, 256, 0, stream>>>(Wv, wvp);
    pack_w<<<blocks, 256, 0, stream>>>(Wo, wop);
  }
  // 3. QKV projections (32 rows per block now)
  {
    dim3 grid((NB * SQ) / 32, 2);
    gemm_f16<<<grid, 128, 0, stream>>>(xh, wqp, bq,  qh, nullptr, 0);
    gemm_f16<<<grid, 128, 0, stream>>>(xh, wkp, bk,  kh, nullptr, 0);
    gemm_f16<<<grid, 128, 0, stream>>>(xh, wvp, bvp, vt, nullptr, 1);
  }
  // 4. attention
  attn_kernel<<<NB * NH, 128, 0, stream>>>(qh, kh, vt, hm, rb, ctx);
  // 5. output projection -> f32 out
  {
    dim3 grid((NB * SQ) / 32, 2);
    gemm_f16<<<grid, 128, 0, stream>>>(ctx, wop, bo, nullptr, out, 2);
  }
  (void)in_sizes; (void)n_in; (void)out_size; (void)ws_size;
}
